// MySAINT_28252294873368
// MI455X (gfx1250) — compile-verified, matrix operand order checked by reading, not run on hardware
//
#include <hip/hip_runtime.h>
#include <hip/hip_bf16.h>
#include <math.h>

// ---------------------------------------------------------------------------
// SAINT-style tabular transformer for MI455X (gfx1250, wave32, WMMA).
// All GEMMs use v_wmma_f32_16x16x32_f16 (f16 in, f32 accumulate).
// Problem fits in L2 (192MB) -> latency/compute bound; few fused kernels.
// Tile index math is forced wave-uniform via readfirstlane -> SALU.
// ---------------------------------------------------------------------------

#define B_    1024
#define F_    100
#define DIM   32
#define DEPTH 6
#define HEADS 8
#define DH    16
#define INNER 128
#define HID   100
#define FFH   128
#define FYH   1000
#define S_    101
#define SP    112           // seq padded to 7*16
#define SKP   128           // K-dim pad for P@V (4 * 32)
#define M_    (B_ * SP)     // 114688 rows in the flattened token stream
#define LN_EPS 1e-5f

typedef _Float16 half8  __attribute__((ext_vector_type(8)));
typedef _Float16 half16 __attribute__((ext_vector_type(16)));
typedef float    float8 __attribute__((ext_vector_type(8)));

__device__ __forceinline__ int rfl(int v) { return __builtin_amdgcn_readfirstlane(v); }

// ---- WMMA helpers ----------------------------------------------------------

__device__ __forceinline__ float8 wmma32(half16 a, half16 b, float8 c) {
  return __builtin_amdgcn_wmma_f32_16x16x32_f16(
      /*neg_a=*/false, a, /*neg_b=*/false, b,
      /*c_mod=*/(short)0, c, /*reuse_a=*/false, /*reuse_b=*/false);
}

// A/BT fragment, row-major f16 matrix, full K=32 step at column k0.
// lane<16 holds K {0..7,16..23}; lane>=16 holds K {8..15,24..31} -> two 16B loads.
__device__ __forceinline__ half16 frag_k32(const _Float16* base, int row, int ld,
                                           int k0, int lane) {
  int hi = (lane >> 4) & 1;
  const _Float16* p = base + (size_t)row * ld + k0 + hi * 8;
  half8 lo  = *(const half8*)p;
  half8 hi8 = *(const half8*)(p + 16);
  half16 a;
#pragma unroll
  for (int i = 0; i < 8; i++) { a[i] = lo[i]; a[i + 8] = hi8[i]; }
  return a;
}

// Fragment for a matrix whose real K is only 16 (q/k, DH=16): upper K zero.
__device__ __forceinline__ half16 frag_k16(const _Float16* base, int row, int lane) {
  int hi = (lane >> 4) & 1;
  half8 lo = *(const half8*)(base + (size_t)row * 16 + hi * 8);
  half16 a;
#pragma unroll
  for (int i = 0; i < 8; i++) { a[i] = lo[i]; a[i + 8] = (_Float16)0.f; }
  return a;
}

// ---- weight conversion (f32 -> f16, transposed to N x K) -------------------

#define CVT_N1 (DEPTH * 384 * 32)   // wqkvT
#define CVT_N2 (DEPTH * 32 * 128)   // woutT / ffw1T / ffw2T (same count)
#define CVT_N5 (1008 * 32)          // fyw1T (N padded to 1008)
#define CVT_N6 (B_ * HEADS * DH * 16) // zero vT pad cols [112,128)
#define CVT_TOTAL (CVT_N1 + 3 * CVT_N2 + CVT_N5 + CVT_N6)

__global__ __launch_bounds__(256) void k_convert(
    const float* wqkv, const float* wout, const float* ffw1, const float* ffw2,
    const float* fyw1, _Float16* wqkvT, _Float16* woutT, _Float16* ffw1T,
    _Float16* ffw2T, _Float16* fyw1T, _Float16* vT) {
  int i = blockIdx.x * 256 + threadIdx.x;
  if (i < CVT_N1) {  // wqkvT[d][n<384][k<32] = wqkv[d][k][n]
    int d = i / 12288, r = i % 12288, n = r / 32, k = r % 32;
    wqkvT[i] = (_Float16)wqkv[d * 12288 + k * 384 + n];
    return;
  }
  i -= CVT_N1;
  if (i < CVT_N2) {  // woutT[d][n<32][k<128] = wout[d][k][n]
    int d = i / 4096, r = i % 4096, n = r / 128, k = r % 128;
    woutT[i] = (_Float16)wout[d * 4096 + k * 32 + n];
    return;
  }
  i -= CVT_N2;
  if (i < CVT_N2) {  // ffw1T[d][n<128][k<32] = ff_w1[d][k][n]
    int d = i / 4096, r = i % 4096, n = r / 32, k = r % 32;
    ffw1T[i] = (_Float16)ffw1[d * 4096 + k * 128 + n];
    return;
  }
  i -= CVT_N2;
  if (i < CVT_N2) {  // ffw2T[d][n<32][k<128] = ff_w2[d][k][n]
    int d = i / 4096, r = i % 4096, n = r / 128, k = r % 128;
    ffw2T[i] = (_Float16)ffw2[d * 4096 + k * 32 + n];
    return;
  }
  i -= CVT_N2;
  if (i < CVT_N5) {  // fyw1T[n<1008][k<32] = fy_w1[k][n] (zero pad n>=1000)
    int n = i / 32, k = i % 32;
    fyw1T[i] = (_Float16)((n < FYH) ? fyw1[k * FYH + n] : 0.f);
    return;
  }
  i -= CVT_N5;
  if (i < CVT_N6) {  // zero vT columns j in [112,128)
    int r = i / 16, cj = SP + (i % 16);
    vT[(size_t)r * SKP + cj] = (_Float16)0.f;
  }
}

// ---- feature encoder (per-feature MLP + mask/cat embeddings) ---------------

__global__ __launch_bounds__(128) void k_encoder(
    const float* x, const float* w1, const float* b1, const float* w2,
    const float* b2, const float* mask_emb, const float* cat_emb, float* z) {
  int gw   = rfl((blockIdx.x * 128 + threadIdx.x) >> 5);  // wave id (scalar)
  int lane = threadIdx.x & 31;                            // lane == output dim
  if (gw < B_ * F_) {
    int b = gw / F_, f = gw % F_;
    float xv = x[(size_t)b * (F_ + 1) + 1 + f];
    float enc;
    if (__builtin_amdgcn_readfirstlane(xv != xv ? 1 : 0)) {  // NaN (wave-uniform)
      enc = mask_emb[(size_t)(2 * f) * DIM + lane];
    } else {
      float acc = 0.f;
      const float* w1f = w1 + (size_t)f * HID;
      const float* b1f = b1 + (size_t)f * HID;
      const float* w2f = w2 + (size_t)f * HID * DIM;
      for (int h = 0; h < HID; h++) {
        float hv = fmaxf(xv * w1f[h] + b1f[h], 0.f);
        acc += hv * w2f[(size_t)h * DIM + lane];
      }
      enc = acc + b2[(size_t)f * DIM + lane];
    }
    z[((size_t)b * SP + 1 + f) * DIM + lane] = enc;
  } else if (gw < B_ * F_ + B_) {
    int b = gw - B_ * F_;
    int ci = rfl((int)x[(size_t)b * (F_ + 1)]);
    z[((size_t)b * SP) * DIM + lane] = cat_emb[(size_t)ci * DIM + lane];
    for (int s = S_; s < SP; s++)
      z[((size_t)b * SP + s) * DIM + lane] = 0.f;   // pad rows
  }
}

// ---- LayerNorm over DIM=32: one wave per token row -------------------------

__global__ __launch_bounds__(128) void k_layernorm(const float* z, const float* g,
                                                   const float* b, _Float16* yh) {
  int row  = rfl((blockIdx.x * 128 + threadIdx.x) >> 5);
  int lane = threadIdx.x & 31;
  if (row >= M_) return;
  float v = z[(size_t)row * DIM + lane];
  float s = v;
#pragma unroll
  for (int m = 16; m > 0; m >>= 1) s += __shfl_xor(s, m, 32);
  float mean = s * (1.f / 32.f);
  float d = v - mean;
  float q = d * d;
#pragma unroll
  for (int m = 16; m > 0; m >>= 1) q += __shfl_xor(q, m, 32);
  float var = q * (1.f / 32.f);
  float y = d * rsqrtf(var + LN_EPS) * g[lane] + b[lane];
  yh[(size_t)row * DIM + lane] = (_Float16)y;
}

// ---- qkv GEMM: (M,32) @ (32,384), scatter into q/k/vT ----------------------
// All scatter-target selection is wave-uniform (tiles are 16 wide, SP = 7*16).

__global__ __launch_bounds__(128) void k_gemm_qkv(const _Float16* yh,
                                                  const _Float16* wT, _Float16* q,
                                                  _Float16* k, _Float16* vT) {
  int gw   = rfl((blockIdx.x * 128 + threadIdx.x) >> 5);
  int lane = threadIdx.x & 31;
  const int Nt = 384 / 16;
  if (gw >= (M_ / 16) * Nt) return;
  int m0 = (gw / Nt) * 16, n0 = (gw % Nt) * 16;
  half16 af = frag_k32(yh, m0 + (lane & 15), DIM, 0, lane);
  half16 bf = frag_k32(wT, n0 + (lane & 15), DIM, 0, lane);
  float8 c = {};
  c = wmma32(af, bf, c);
  int d  = lane & 15;
  int hi = (lane >> 4) & 1;
  int nt = n0 >> 4;          // scalar: 0..23 -> q heads 0..7, k heads, v heads
  int b  = m0 / SP;          // scalar (m0 multiple of 16, 7 tiles per row)
  int s0 = m0 % SP + hi * 8; // + per-lane-half offset
  if (nt < 8) {
    int h = nt;
    _Float16* dst = q + (((size_t)(b * HEADS + h)) * SP + s0) * DH + d;
#pragma unroll
    for (int r = 0; r < 8; r++) dst[r * DH] = (_Float16)c[r];
  } else if (nt < 16) {
    int h = nt - 8;
    _Float16* dst = k + (((size_t)(b * HEADS + h)) * SP + s0) * DH + d;
#pragma unroll
    for (int r = 0; r < 8; r++) dst[r * DH] = (_Float16)c[r];
  } else {
    int h = nt - 16;
    // V transposed: consecutive r are contiguous in s -> one b128 store/lane
    _Float16* dst = vT + (((size_t)(b * HEADS + h)) * DH + d) * SKP + s0;
    half8 v8;
#pragma unroll
    for (int r = 0; r < 8; r++) v8[r] = (_Float16)c[r];
    *(half8*)dst = v8;
  }
}

// ---- attention: one workgroup (4 waves) per (b,h) --------------------------

__global__ __launch_bounds__(128) void k_attn(const _Float16* q, const _Float16* k,
                                              const _Float16* vT, _Float16* oh) {
  __shared__ float lg[SP][SP];  // 112x112 f32 logits, reused for P (49KB LDS)
  int bh = blockIdx.x, b = bh >> 3, h = bh & 7;
  int tid = threadIdx.x, lane = tid & 31, w = rfl(tid >> 5);
  const _Float16* qb = q + (size_t)bh * SP * DH;
  const _Float16* kb = k + (size_t)bh * SP * DH;
  // phase 1: logits = (Q @ K^T) * DH^-0.5   (49 tiles of 16x16, K=16 zero-padded)
  for (int t = w; t < 49; t += 4) {
    int ti = t / 7, tj = t % 7;
    half16 af = frag_k16(qb, ti * 16 + (lane & 15), lane);
    half16 bf = frag_k16(kb, tj * 16 + (lane & 15), lane);
    float8 c = {};
    c = wmma32(af, bf, c);
    int nn = tj * 16 + (lane & 15);
    int mb = ti * 16 + (((lane >> 4) & 1) << 3);
#pragma unroll
    for (int r = 0; r < 8; r++) lg[mb + r][nn] = c[r] * 0.25f;  // 1/sqrt(16)
  }
  __syncthreads();
  // phase 2: row softmax over the 101 real key tokens; zero the pad columns
  if (tid < SP) {
    float mx = -3.4e38f;
    for (int j = 0; j < S_; j++) mx = fmaxf(mx, lg[tid][j]);
    float sum = 0.f;
    for (int j = 0; j < S_; j++) { float e = expf(lg[tid][j] - mx); lg[tid][j] = e; sum += e; }
    float inv = 1.f / sum;
    for (int j = 0; j < S_; j++) lg[tid][j] *= inv;
    for (int j = S_; j < SP; j++) lg[tid][j] = 0.f;
  }
  __syncthreads();
  // phase 3: O = P @ V   (A from LDS, B^T = vT from global, K padded to 128)
  const _Float16* vb = vT + (size_t)bh * DH * SKP;
  int lh = lane & 15, hi = (lane >> 4) & 1;
  for (int ti = w; ti < 7; ti += 4) {
    float8 c = {};
    int row = ti * 16 + lh;
#pragma unroll
    for (int k0 = 0; k0 < SKP; k0 += 32) {
      half16 af;
#pragma unroll
      for (int i = 0; i < 16; i++) {
        int kk = k0 + ((i < 8) ? i : i + 8) + hi * 8;
        af[i] = (kk < SP) ? (_Float16)lg[row][kk] : (_Float16)0.f;
      }
      half16 bf = frag_k32(vb, lh, SKP, k0, lane);  // row lh = head-dim d
      c = wmma32(af, bf, c);
    }
    int s0 = ti * 16 + (hi << 3);
    _Float16* dst = oh + ((size_t)b * SP + s0) * INNER + h * DH + lh;
#pragma unroll
    for (int r = 0; r < 8; r++) dst[(size_t)r * INNER] = (_Float16)c[r];
  }
}

// ---- GEMM (M,128)@(128,32) + bias + residual(y) -> z (f32), zero pad rows --

__global__ __launch_bounds__(128) void k_gemm_addz(const _Float16* A,
                                                   const _Float16* BT,
                                                   const float* bias,
                                                   const _Float16* y, float* z) {
  int gw   = rfl((blockIdx.x * 128 + threadIdx.x) >> 5);
  int lane = threadIdx.x & 31;
  const int Nt = 2;  // N = 32
  if (gw >= (M_ / 16) * Nt) return;
  int m0 = (gw / Nt) * 16, n0 = (gw % Nt) * 16;
  float8 c = {};
#pragma unroll
  for (int k0 = 0; k0 < 128; k0 += 32) {
    half16 af = frag_k32(A, m0 + (lane & 15), 128, k0, lane);
    half16 bf = frag_k32(BT, n0 + (lane & 15), 128, k0, lane);
    c = wmma32(af, bf, c);
  }
  int nn = n0 + (lane & 15);
  int hi = (lane >> 4) & 1;
  int s0 = m0 % SP + (hi << 3);   // scalar row-in-seq base (+ half offset)
  int mb = m0 + (hi << 3);
  float bv = bias[nn];
  const _Float16* ysrc = y + (size_t)mb * DIM + nn;
  float* dst = z + (size_t)mb * DIM + nn;
#pragma unroll
  for (int r = 0; r < 8; r++) {
    dst[(size_t)r * DIM] =
        (s0 + r < S_) ? ((float)ysrc[(size_t)r * DIM] + c[r] + bv) : 0.f;
  }
}

// ---- FFN first GEMM (M,32)@(32,128) + bias + exact GELU -> f16 -------------

__global__ __launch_bounds__(128) void k_gemm_ffh(const _Float16* A,
                                                  const _Float16* BT,
                                                  const float* bias, _Float16* o) {
  int gw   = rfl((blockIdx.x * 128 + threadIdx.x) >> 5);
  int lane = threadIdx.x & 31;
  const int Nt = 8;  // N = 128
  if (gw >= (M_ / 16) * Nt) return;
  int m0 = (gw / Nt) * 16, n0 = (gw % Nt) * 16;
  half16 af = frag_k32(A, m0 + (lane & 15), DIM, 0, lane);
  half16 bf = frag_k32(BT, n0 + (lane & 15), DIM, 0, lane);
  float8 c = {};
  c = wmma32(af, bf, c);
  int nn = n0 + (lane & 15);
  int mb = m0 + (((lane >> 4) & 1) << 3);
  float bv = bias[nn];
  _Float16* dst = o + (size_t)mb * FFH + nn;
#pragma unroll
  for (int r = 0; r < 8; r++) {
    float v = c[r] + bv;
    float g = 0.5f * v * (1.f + erff(v * 0.70710678118f));  // exact GELU
    dst[(size_t)r * FFH] = (_Float16)g;
  }
}

// ---- head GEMM 1: relu(z[:,0,:] @ fy_w1 + b1), N padded to 1008 ------------

__global__ __launch_bounds__(128) void k_head1(const float* z, const _Float16* w1T,
                                               const float* b1, _Float16* hh) {
  int gw   = rfl((blockIdx.x * 128 + threadIdx.x) >> 5);
  int lane = threadIdx.x & 31;
  const int Nt = 1008 / 16;
  if (gw >= (B_ / 16) * Nt) return;
  int m0 = (gw / Nt) * 16, n0 = (gw % Nt) * 16;
  int lh = lane & 15, hi = (lane >> 4) & 1;
  half16 af;
  int row = m0 + lh;  // batch index
#pragma unroll
  for (int i = 0; i < 16; i++) {
    int kk = ((i < 8) ? i : i + 8) + hi * 8;
    af[i] = (_Float16)z[((size_t)row * SP) * DIM + kk];  // token 0 row, f32->f16
  }
  half16 bf = frag_k32(w1T, n0 + lh, DIM, 0, lane);
  float8 c = {};
  c = wmma32(af, bf, c);
  int nn = n0 + lh;
  int mb = m0 + (hi << 3);
  float bv = (nn < FYH) ? b1[nn] : 0.f;
  _Float16* dst = hh + (size_t)mb * 1008 + nn;
#pragma unroll
  for (int r = 0; r < 8; r++) {
    float v = fmaxf(c[r] + bv, 0.f);
    dst[(size_t)r * 1008] = (_Float16)v;
  }
}

// ---- head GEMM 2: (B,1000) . (1000,) dot per batch row ---------------------

__global__ __launch_bounds__(128) void k_head2(const _Float16* hh, const float* w2,
                                               const float* b2, float* out) {
  int gw   = rfl((blockIdx.x * 128 + threadIdx.x) >> 5);
  int lane = threadIdx.x & 31;
  if (gw >= B_) return;
  float acc = 0.f;
  for (int n = lane; n < FYH; n += 32)
    acc += (float)hh[(size_t)gw * 1008 + n] * w2[n];
#pragma unroll
  for (int m = 16; m > 0; m >>= 1) acc += __shfl_xor(acc, m, 32);
  if (lane == 0) out[gw] = acc + b2[0];
}

// ---------------------------------------------------------------------------

extern "C" void kernel_launch(void* const* d_in, const int* in_sizes, int n_in,
                              void* d_out, int out_size, void* d_ws, size_t ws_size,
                              hipStream_t stream) {
  const float* x        = (const float*)d_in[0];
  const float* mlp_w1   = (const float*)d_in[1];
  const float* mlp_b1   = (const float*)d_in[2];
  const float* mlp_w2   = (const float*)d_in[3];
  const float* mlp_b2   = (const float*)d_in[4];
  const float* mask_emb = (const float*)d_in[5];
  const float* cat_emb  = (const float*)d_in[6];
  const float* ln1_g    = (const float*)d_in[7];
  const float* ln1_b    = (const float*)d_in[8];
  const float* wqkv     = (const float*)d_in[9];
  const float* wout     = (const float*)d_in[10];
  const float* bout     = (const float*)d_in[11];
  const float* ln2_g    = (const float*)d_in[12];
  const float* ln2_b    = (const float*)d_in[13];
  const float* ff_w1    = (const float*)d_in[14];
  const float* ff_b1    = (const float*)d_in[15];
  const float* ff_w2    = (const float*)d_in[16];
  const float* ff_b2    = (const float*)d_in[17];
  const float* fy_w1    = (const float*)d_in[18];
  const float* fy_b1    = (const float*)d_in[19];
  const float* fy_w2    = (const float*)d_in[20];
  const float* fy_b2    = (const float*)d_in[21];
  float* out = (float*)d_out;

  // workspace carve (aligned 256B); total ~140MB
  char* ws = (char*)d_ws;
  size_t o = 0;
  auto carve = [&](size_t bytes) -> char* {
    char* p = ws + o;
    o += (bytes + 255) & ~(size_t)255;
    return p;
  };
  float*    z     = (float*)carve((size_t)M_ * DIM * 4);
  _Float16* yh    = (_Float16*)carve((size_t)M_ * DIM * 2);
  _Float16* qh    = (_Float16*)carve((size_t)B_ * HEADS * SP * DH * 2);
  _Float16* kh    = (_Float16*)carve((size_t)B_ * HEADS * SP * DH * 2);
  _Float16* vT    = (_Float16*)carve((size_t)B_ * HEADS * DH * SKP * 2);
  _Float16* oh    = (_Float16*)carve((size_t)M_ * INNER * 2);  // reused as FFN hidden
  _Float16* wqkvT = (_Float16*)carve((size_t)DEPTH * 384 * 32 * 2);
  _Float16* woutT = (_Float16*)carve((size_t)DEPTH * 32 * 128 * 2);
  _Float16* ffw1T = (_Float16*)carve((size_t)DEPTH * 128 * 32 * 2);
  _Float16* ffw2T = (_Float16*)carve((size_t)DEPTH * 32 * 128 * 2);
  _Float16* fyw1T = (_Float16*)carve((size_t)1008 * 32 * 2);
  _Float16* hh    = (_Float16*)carve((size_t)B_ * 1008 * 2);

  k_convert<<<(CVT_TOTAL + 255) / 256, 256, 0, stream>>>(
      wqkv, wout, ff_w1, ff_w2, fy_w1, wqkvT, woutT, ffw1T, ffw2T, fyw1T, vT);

  k_encoder<<<(B_ * F_ + B_) / 4, 128, 0, stream>>>(
      x, mlp_w1, mlp_b1, mlp_w2, mlp_b2, mask_emb, cat_emb, z);

  for (int l = 0; l < DEPTH; l++) {
    k_layernorm<<<M_ / 4, 128, 0, stream>>>(z, ln1_g + l * 32, ln1_b + l * 32, yh);
    k_gemm_qkv<<<(M_ / 16) * 24 / 4, 128, 0, stream>>>(yh, wqkvT + l * 12288, qh, kh, vT);
    k_attn<<<B_ * HEADS, 128, 0, stream>>>(qh, kh, vT, oh);
    k_gemm_addz<<<(M_ / 16) * 2 / 4, 128, 0, stream>>>(oh, woutT + l * 4096,
                                                       bout + l * 32, yh, z);
    k_layernorm<<<M_ / 4, 128, 0, stream>>>(z, ln2_g + l * 32, ln2_b + l * 32, yh);
    k_gemm_ffh<<<(M_ / 16) * 8 / 4, 128, 0, stream>>>(yh, ffw1T + l * 4096,
                                                      ff_b1 + l * 128, oh);
    k_gemm_addz<<<(M_ / 16) * 2 / 4, 128, 0, stream>>>(oh, ffw2T + l * 4096,
                                                       ff_b2 + l * 32, yh, z);
  }

  k_head1<<<(B_ / 16) * (1008 / 16) / 4, 128, 0, stream>>>(z, fyw1T, fy_b1, hh);
  k_head2<<<B_ / 4, 128, 0, stream>>>(hh, fy_w2, fy_b2, out);

  (void)in_sizes; (void)n_in; (void)out_size; (void)ws_size;
}
